// ObjectContext_38190849196084
// MI455X (gfx1250) — compile-verified, matrix operand order checked by reading, not run on hardware
//
#include <hip/hip_runtime.h>

// ---------------------------------------------------------------------------
// Types for CDNA5 WMMA (wave32)
// ---------------------------------------------------------------------------
typedef __attribute__((ext_vector_type(16))) __bf16        bf16x16;
typedef __attribute__((ext_vector_type(8)))  float         f32x8;
typedef __attribute__((ext_vector_type(4)))  unsigned int  u32x4;
typedef __attribute__((ext_vector_type(8)))  unsigned int  u32x8;

#define NBOX   512
#define BIMG   16
#define TSEQ   32
#define FEAT   2048
#define NCLS   80
#define CTX    1024
#define INDIM  3152
#define KPAD   3168          // INDIM padded to a multiple of 32 for WMMA K-steps
#define FC     1024
#define H      1024
#define G4H    4096

// fp32 -> bf16, round-to-nearest-even
__device__ __forceinline__ unsigned short f2bf(float x) {
    unsigned int u = __builtin_bit_cast(unsigned int, x);
    unsigned int r = u + 0x7FFFu + ((u >> 16) & 1u);
    return (unsigned short)(r >> 16);
}

__device__ __forceinline__ float sigmoidf_(float x) {
    return 1.0f / (1.0f + expf(-x));
}

// Load one 16x32 bf16 fragment for this lane: two contiguous 16B loads.
// p0 points at (row*ld + kbase + 8*kh) in elements; second run is +16 elems.
__device__ __forceinline__ bf16x16 load_frag(const unsigned short* p0) {
    u32x4 lo = *reinterpret_cast<const u32x4*>(p0);
    u32x4 hi = *reinterpret_cast<const u32x4*>(p0 + 16);
    u32x8 c = {lo[0], lo[1], lo[2], lo[3], hi[0], hi[1], hi[2], hi[3]};
    return __builtin_bit_cast(bf16x16, c);
}

// Register-blocked MTxNT 16x16-tile block of A[M,K] @ B[N,K]^T.
// A fragments are reused across NT columns, B fragments across MT rows:
// per K-step, MT+NT fragment loads feed MT*NT WMMAs.
template<int MT, int NT>
__device__ __forceinline__ void wmma_block(
    const unsigned short* __restrict__ A, int lda,
    const unsigned short* __restrict__ B, int ldb,
    int arow0, int brow0, int K, int lane, f32x8 (&acc)[MT][NT])
{
    const int kh = lane >> 4;
    const int mr = lane & 15;
    const unsigned short* ap[MT];
    const unsigned short* bp[NT];
#pragma unroll
    for (int i = 0; i < MT; ++i)
        ap[i] = A + (size_t)(arow0 + 16 * i + mr) * lda + 8 * kh;
#pragma unroll
    for (int j = 0; j < NT; ++j)
        bp[j] = B + (size_t)(brow0 + 16 * j + mr) * ldb + 8 * kh;

    for (int k = 0; k < K; k += 32) {
        bf16x16 a[MT], b[NT];
#pragma unroll
        for (int i = 0; i < MT; ++i) {
            __builtin_prefetch(ap[i] + k + 512, 0, 0);   // global_prefetch_b8
            a[i] = load_frag(ap[i] + k);
        }
#pragma unroll
        for (int j = 0; j < NT; ++j) {
            __builtin_prefetch(bp[j] + k + 512, 0, 0);
            b[j] = load_frag(bp[j] + k);
        }
#pragma unroll
        for (int i = 0; i < MT; ++i)
#pragma unroll
            for (int j = 0; j < NT; ++j)
                acc[i][j] = __builtin_amdgcn_wmma_f32_16x16x32_bf16(
                    false, a[i], false, b[j], (short)0, acc[i][j], false, false);
    }
}

// ---------------------------------------------------------------------------
// Preprocessing kernels (one-shot converts; bandwidth trivial vs. GEMMs)
// ---------------------------------------------------------------------------
__global__ void f32_to_bf16_kernel(const float* __restrict__ src,
                                   unsigned short* __restrict__ dst, int n) {
    int i = blockIdx.x * blockDim.x + threadIdx.x;
    if (i < n) dst[i] = f2bf(src[i]);
}

// w_fc [1024, 3152] -> bf16 [1024, KPAD] (zero padded K)
__global__ void convert_wfc_kernel(const float* __restrict__ src,
                                   unsigned short* __restrict__ dst) {
    int i = blockIdx.x * blockDim.x + threadIdx.x;
    if (i >= 1024 * KPAD) return;
    int row = i / KPAD, col = i - row * KPAD;
    float v = (col < INDIM) ? src[row * INDIM + col] : 0.0f;
    dst[i] = f2bf(v);
}

// X bf16 [512, KPAD] = concat(box_feats, boxes_ext[:,5:], ctx[box_im_ids]), pad 0
__global__ void build_x_kernel(const float* __restrict__ box_feats,
                               const float* __restrict__ boxes_ext,
                               const float* __restrict__ spatial_ctx,
                               const int*   __restrict__ box_im_ids,
                               unsigned short* __restrict__ X) {
    int i = blockIdx.x * blockDim.x + threadIdx.x;
    if (i >= NBOX * KPAD) return;
    int n = i / KPAD, c = i - n * KPAD;
    float v;
    if (c < FEAT)                 v = box_feats[n * FEAT + c];
    else if (c < FEAT + NCLS)     v = boxes_ext[n * (5 + NCLS) + 5 + (c - FEAT)];
    else if (c < INDIM)           v = spatial_ctx[box_im_ids[n] * CTX + (c - FEAT - NCLS)];
    else                          v = 0.0f;
    X[i] = f2bf(v);
}

// ---------------------------------------------------------------------------
// Stage 1: emb = relu(X @ Wfc^T + b_fc)   M=512 N=1024 K=KPAD
// 2x4 register blocking: 256 waves, 32-row x 64-col macro tile per wave.
// launch_bounds(256,1): allow a full VGPR allocation (no accumulator spills).
// ---------------------------------------------------------------------------
__global__ void __launch_bounds__(256, 1)
gemm_emb_kernel(const unsigned short* __restrict__ X,
                const unsigned short* __restrict__ Wfc,
                const float* __restrict__ b_fc,
                float* __restrict__ emb_f32,
                unsigned short* __restrict__ emb_bf16) {
    int wave = (blockIdx.x * blockDim.x + threadIdx.x) >> 5;
    int lane = threadIdx.x & 31;
    int mblk = wave >> 4;               // 16 macro rows (of 32)
    int nblk = wave & 15;               // 16 macro cols (of 64)
    int m0 = mblk * 32, n0 = nblk * 64;
    f32x8 acc[2][4] = {};
    wmma_block<2, 4>(X, KPAD, Wfc, KPAD, m0, n0, KPAD, lane, acc);
    int kh = lane >> 4, nc = lane & 15;
#pragma unroll
    for (int j = 0; j < 4; ++j) {
        int n = n0 + 16 * j + nc;
        float bias = b_fc[n];
#pragma unroll
        for (int i = 0; i < 2; ++i)
#pragma unroll
            for (int r = 0; r < 8; ++r) {
                int m = m0 + 16 * i + r + 8 * kh;
                float v = acc[i][j][r] + bias;
                v = v > 0.0f ? v : 0.0f;
                emb_f32[m * FC + n]  = v;
                emb_bf16[m * FC + n] = f2bf(v);
            }
    }
}

// ---------------------------------------------------------------------------
// Stage 2: xg[dir] = emb @ w_ih[dir]^T + (b_ih+b_hh)  M=512 N=4096 K=1024
// 2x4 register blocking: 2048 waves. launch_bounds(256,1) to avoid spills.
// ---------------------------------------------------------------------------
__global__ void __launch_bounds__(256, 1)
gemm_xg_kernel(const unsigned short* __restrict__ emb_bf16,
               const unsigned short* __restrict__ wih_f,
               const unsigned short* __restrict__ wih_r,
               const float* __restrict__ bih_f, const float* __restrict__ bhh_f,
               const float* __restrict__ bih_r, const float* __restrict__ bhh_r,
               float* __restrict__ xg_f, float* __restrict__ xg_r) {
    int wave = (blockIdx.x * blockDim.x + threadIdx.x) >> 5;
    int lane = threadIdx.x & 31;
    int dir  = wave >> 10;              // 1024 waves per direction
    int rem  = wave & 1023;
    int mblk = rem >> 6;                // 16 macro rows (of 32)
    int nblk = rem & 63;                // 64 macro cols (of 64)
    int m0 = mblk * 32, n0 = nblk * 64;
    const unsigned short* W = dir ? wih_r : wih_f;
    const float* bi = dir ? bih_r : bih_f;
    const float* bh = dir ? bhh_r : bhh_f;
    float* xg = dir ? xg_r : xg_f;
    f32x8 acc[2][4] = {};
    wmma_block<2, 4>(emb_bf16, FC, W, FC, m0, n0, FC, lane, acc);
    int kh = lane >> 4, nc = lane & 15;
#pragma unroll
    for (int j = 0; j < 4; ++j) {
        int n = n0 + 16 * j + nc;
        float bias = bi[n] + bh[n];
#pragma unroll
        for (int i = 0; i < 2; ++i)
#pragma unroll
            for (int r = 0; r < 8; ++r) {
                int m = m0 + 16 * i + r + 8 * kh;
                xg[(size_t)m * G4H + n] = acc[i][j][r] + bias;
            }
    }
}

// ---------------------------------------------------------------------------
// Recurrence init: zero h(bf16), c, hsum for both directions
// ---------------------------------------------------------------------------
__global__ void init_state_kernel(unsigned short* __restrict__ h_bf16,
                                  float* __restrict__ c_state,
                                  float* __restrict__ hsum) {
    int i = blockIdx.x * blockDim.x + threadIdx.x;
    if (i >= 2 * BIMG * H) return;
    h_bf16[i] = 0;
    c_state[i] = 0.0f;
    hsum[i] = 0.0f;
}

// ---------------------------------------------------------------------------
// One recurrent step, both directions: g = xg_t + h @ w_hh^T  (M=16 N=4096)
// 1x2 N-blocking (A reused across both tiles), 256 waves so the L2-resident
// w_hh stream keeps many concurrent requesters. Default occupancy bounds.
// ---------------------------------------------------------------------------
__global__ void lstm_step_gemm_kernel(const unsigned short* __restrict__ h_bf16,
                                      const unsigned short* __restrict__ whh_f,
                                      const unsigned short* __restrict__ whh_r,
                                      const float* __restrict__ xg_f,
                                      const float* __restrict__ xg_r,
                                      float* __restrict__ gbuf, int t) {
    int wave = (blockIdx.x * blockDim.x + threadIdx.x) >> 5;
    int lane = threadIdx.x & 31;
    int dir  = wave >> 7;               // 128 macro cols (of 32) per direction
    int nblk = wave & 127;
    int n0   = nblk * 32;
    const unsigned short* A = h_bf16 + dir * BIMG * H;
    const unsigned short* W = dir ? whh_r : whh_f;
    const float* xg = dir ? xg_r : xg_f;
    int tt = dir ? (TSEQ - 1 - t) : t;  // reverse direction walks backwards
    f32x8 acc[1][2] = {};
    wmma_block<1, 2>(A, H, W, H, 0, n0, H, lane, acc);
    int kh = lane >> 4, nc = lane & 15;
#pragma unroll
    for (int j = 0; j < 2; ++j) {
        int n = n0 + 16 * j + nc;
#pragma unroll
        for (int r = 0; r < 8; ++r) {
            int b = r + 8 * kh;         // batch row 0..15
            gbuf[dir * BIMG * G4H + b * G4H + n] =
                acc[0][j][r] + xg[(size_t)(b * TSEQ + tt) * G4H + n];
        }
    }
}

// Gate nonlinearity + state update + running time-sum (mean is order-invariant)
__global__ void lstm_gates_kernel(const float* __restrict__ gbuf,
                                  float* __restrict__ c_state,
                                  float* __restrict__ hsum,
                                  unsigned short* __restrict__ h_bf16) {
    int i = blockIdx.x * blockDim.x + threadIdx.x;
    if (i >= 2 * BIMG * H) return;
    int dir = i >> 14;                  // 16*1024 = 16384 per direction
    int rem = i & 16383;
    int b   = rem >> 10;
    int j   = rem & 1023;
    const float* g = gbuf + dir * BIMG * G4H + b * G4H;
    float ig = g[j];
    float fg = g[j + H];
    float gg = g[j + 2 * H];
    float og = g[j + 3 * H];
    float c  = sigmoidf_(fg) * c_state[i] + sigmoidf_(ig) * tanhf(gg);
    float h  = sigmoidf_(og) * tanhf(c);
    c_state[i] = c;
    hsum[i]   += h;
    h_bf16[i]  = f2bf(h);
}

// object_context[b, 0:1024]=mean fwd, [b, 1024:2048]=mean rev
__global__ void finalize_kernel(const float* __restrict__ hsum,
                                float* __restrict__ out) {
    int i = blockIdx.x * blockDim.x + threadIdx.x;
    if (i >= BIMG * 2 * H) return;
    int b   = i >> 11;                  // / 2048
    int j   = i & 2047;
    int dir = j >> 10;
    int jj  = j & 1023;
    out[i] = hsum[dir * BIMG * H + b * H + jj] * (1.0f / TSEQ);
}

// ---------------------------------------------------------------------------
extern "C" void kernel_launch(void* const* d_in, const int* in_sizes, int n_in,
                              void* d_out, int out_size, void* d_ws, size_t ws_size,
                              hipStream_t stream) {
    const float* boxes_ext   = (const float*)d_in[0];
    const float* box_feats   = (const float*)d_in[1];
    const float* spatial_ctx = (const float*)d_in[2];
    const int*   box_im_ids  = (const int*)  d_in[4];
    const float* w_fc        = (const float*)d_in[5];
    const float* b_fc        = (const float*)d_in[6];
    const float* w_ih_f      = (const float*)d_in[7];
    const float* w_hh_f      = (const float*)d_in[8];
    const float* b_ih_f      = (const float*)d_in[9];
    const float* b_hh_f      = (const float*)d_in[10];
    const float* w_ih_r      = (const float*)d_in[11];
    const float* w_hh_r      = (const float*)d_in[12];
    const float* b_ih_r      = (const float*)d_in[13];
    const float* b_hh_r      = (const float*)d_in[14];

    float* out_ctx = (float*)d_out;                       // [16, 2048]
    float* out_emb = (float*)d_out + BIMG * 2 * H;        // [512, 1024]

    // --- workspace carve (256B aligned) ---
    char* w = (char*)d_ws;
    auto carve = [&](size_t bytes) {
        void* p = (void*)w;
        w += (bytes + 255) & ~(size_t)255;
        return p;
    };
    unsigned short* Xbf   = (unsigned short*)carve((size_t)NBOX * KPAD * 2);
    unsigned short* Wfc   = (unsigned short*)carve((size_t)1024 * KPAD * 2);
    unsigned short* Wihf  = (unsigned short*)carve((size_t)G4H * FC * 2);
    unsigned short* Wihr  = (unsigned short*)carve((size_t)G4H * FC * 2);
    unsigned short* Whhf  = (unsigned short*)carve((size_t)G4H * H * 2);
    unsigned short* Whhr  = (unsigned short*)carve((size_t)G4H * H * 2);
    unsigned short* embBf = (unsigned short*)carve((size_t)NBOX * FC * 2);
    float* xg_f  = (float*)carve((size_t)NBOX * G4H * 4);
    float* xg_r  = (float*)carve((size_t)NBOX * G4H * 4);
    float* gbuf  = (float*)carve((size_t)2 * BIMG * G4H * 4);
    unsigned short* hBf = (unsigned short*)carve((size_t)2 * BIMG * H * 2);
    float* cSt   = (float*)carve((size_t)2 * BIMG * H * 4);
    float* hSum  = (float*)carve((size_t)2 * BIMG * H * 4);

    const int TB = 256;
    // 1) weight conversion fp32 -> bf16
    convert_wfc_kernel<<<(1024 * KPAD + TB - 1) / TB, TB, 0, stream>>>(w_fc, Wfc);
    f32_to_bf16_kernel<<<(G4H * FC + TB - 1) / TB, TB, 0, stream>>>(w_ih_f, Wihf, G4H * FC);
    f32_to_bf16_kernel<<<(G4H * FC + TB - 1) / TB, TB, 0, stream>>>(w_ih_r, Wihr, G4H * FC);
    f32_to_bf16_kernel<<<(G4H * H  + TB - 1) / TB, TB, 0, stream>>>(w_hh_f, Whhf, G4H * H);
    f32_to_bf16_kernel<<<(G4H * H  + TB - 1) / TB, TB, 0, stream>>>(w_hh_r, Whhr, G4H * H);
    // 2) gather + concat input, bf16
    build_x_kernel<<<(NBOX * KPAD + TB - 1) / TB, TB, 0, stream>>>(
        box_feats, boxes_ext, spatial_ctx, box_im_ids, Xbf);
    // 3) emb GEMM: 256 waves (2x4 blocked) -> 32 blocks
    gemm_emb_kernel<<<32, TB, 0, stream>>>(Xbf, Wfc, b_fc, out_emb, embBf);
    // 4) input-gate GEMMs, both directions: 2048 waves -> 256 blocks
    gemm_xg_kernel<<<256, TB, 0, stream>>>(embBf, Wihf, Wihr,
                                           b_ih_f, b_hh_f, b_ih_r, b_hh_r,
                                           xg_f, xg_r);
    // 5) recurrence
    init_state_kernel<<<(2 * BIMG * H + TB - 1) / TB, TB, 0, stream>>>(hBf, cSt, hSum);
    for (int t = 0; t < TSEQ; ++t) {
        lstm_step_gemm_kernel<<<32, TB, 0, stream>>>(hBf, Whhf, Whhr, xg_f, xg_r, gbuf, t);
        lstm_gates_kernel<<<(2 * BIMG * H + TB - 1) / TB, TB, 0, stream>>>(gbuf, cSt, hSum, hBf);
    }
    // 6) time-mean -> object_context
    finalize_kernel<<<(BIMG * 2 * H + TB - 1) / TB, TB, 0, stream>>>(hSum, out_ctx);
}
